// Transformer_867583394509
// MI455X (gfx1250) — matrix operand order repaired and checked by reading, not compiled, not run on hardware
//
#include <hip/hip_runtime.h>
#include <hip/hip_bf16.h>

typedef __attribute__((ext_vector_type(16))) __bf16 v16bf;
typedef __attribute__((ext_vector_type(2)))  __bf16 bf16x2;
typedef __attribute__((ext_vector_type(8)))  float  f32x8;
typedef __attribute__((ext_vector_type(4)))  unsigned int u32x4;
typedef __attribute__((ext_vector_type(2)))  unsigned int u32x2;
typedef __attribute__((ext_vector_type(4)))  float  f32x4;
typedef unsigned short u16;
typedef unsigned int   u32;

#define TOK   4096      // B*L rows of tokens
#define CDIM  768
#define LSEQ  1024
#define NH    12
#define HDIM  64
#define VOC   50304
#define NLAY  12

#define BM 128
#define BN 128
#define BK 32
#define LDT 40          // LDS row stride in u16 (32 data + 8 pad)

union ABFrag { v16bf f; u32x4 u[2]; };

// native f32 -> bf16 conversion (hardware cvt on gfx1250)
__device__ __forceinline__ u16 f2bf(float x) {
  union { __bf16 b; u16 s; } cv;
  cv.b = (__bf16)x;
  return cv.s;
}
// packed pair convert: lets clang emit a single v_cvt_pk_bf16_f32 lo,hi
__device__ __forceinline__ u32 pack2bf(float lo, float hi) {
  union { bf16x2 v; u32 u; } cv;
  cv.v = bf16x2{(__bf16)lo, (__bf16)hi};
  return cv.u;
}

__device__ __forceinline__ f32x8 wmma_bf16(ABFrag a, ABFrag b, f32x8 c) {
  return __builtin_amdgcn_wmma_f32_16x16x32_bf16(
      /*neg_a=*/false, a.f, /*neg_b=*/false, b.f,
      /*c_mod=*/(short)0, c, /*reuse_a=*/false, /*reuse_b=*/false);
}

// ---------------------------------------------------------------------------
// CDNA5 async global->LDS DMA (tracked by ASYNCcnt). LDS address operand is
// the low 32 bits of the generic pointer (LDS aperture: addr[31:0] = offset).
// ---------------------------------------------------------------------------
__device__ __forceinline__ void async_ld_b128(u16* lds_dst, const u16* gsrc) {
  const u32 laddr = (u32)(size_t)lds_dst;
  asm volatile("global_load_async_to_lds_b128 %0, %1, off"
               :: "v"(laddr), "v"(gsrc)
               : "memory");
}
__device__ __forceinline__ void wait_async0() {
  asm volatile("s_wait_asynccnt 0x0" ::: "memory");
}

// ---------------------------------------------------------------------------
// x[row,c] = wte[idx[row], c] + wpe[row % L, c]
// ---------------------------------------------------------------------------
__global__ __launch_bounds__(256) void embed_kernel(
    const int* __restrict__ idx, const float* __restrict__ wte,
    const float* __restrict__ wpe, float* __restrict__ x)
{
  const size_t i = (size_t)blockIdx.x * 256 + threadIdx.x;  // < TOK*CDIM
  const int c   = (int)(i % CDIM);
  const int row = (int)(i / CDIM);
  const int t   = row & (LSEQ - 1);
  x[i] = wte[(size_t)idx[row] * CDIM + c] + wpe[(size_t)t * CDIM + c];
}

// ---------------------------------------------------------------------------
// LayerNorm: one wave per 768-wide row; writes bf16 normalized activations
// ---------------------------------------------------------------------------
__global__ __launch_bounds__(256) void ln_kernel(
    const float* __restrict__ x, const float* __restrict__ w,
    const float* __restrict__ b, u16* __restrict__ out)
{
  const int row  = blockIdx.x * 8 + (threadIdx.x >> 5);
  const int lane = threadIdx.x & 31;
  const float* xr = x + (size_t)row * CDIM;
  float vals[24];
  float s = 0.f, ss = 0.f;
#pragma unroll
  for (int i = 0; i < 24; ++i) {
    const float v = xr[lane + i * 32];
    vals[i] = v; s += v; ss += v * v;
  }
#pragma unroll
  for (int off = 1; off < 32; off <<= 1) {
    s  += __shfl_xor(s,  off, 32);
    ss += __shfl_xor(ss, off, 32);
  }
  const float mean = s * (1.f / CDIM);
  const float var  = ss * (1.f / CDIM) - mean * mean;
  const float inv  = rsqrtf(var + 1e-5f);
  u16* orow = out + (size_t)row * CDIM;
#pragma unroll
  for (int i = 0; i < 24; ++i) {
    const int c = lane + i * 32;
    orow[c] = f2bf((vals[i] - mean) * inv * w[c] + b[c]);
  }
}

// ---------------------------------------------------------------------------
// WMMA GEMM: out[M,N] = A[M,K](bf16) @ W[N,K]^T(f32->bf16) (+bias)(+gelu)
// mode 0: bf16 store (optional exact GELU); mode 1: fp32 residual +=;
// mode 2: fp32 store (logits).
// Block tile 128x128x32, 8 waves, wave tile 32x64 (2x4 WMMA frags).
// Double-buffered LDS; A tile staged via async global->LDS DMA (no VGPR
// round-trip), B tile register-staged with packed fp32->bf16 conversion.
// ---------------------------------------------------------------------------
__global__ __launch_bounds__(256) void gemm_kernel(
    const u16* __restrict__ A, const float* __restrict__ Bw,
    const float* __restrict__ bias, void* __restrict__ out,
    int M, int N, int K, int mode, int gelu)
{
  __shared__ u16 As[2][BM * LDT];
  __shared__ u16 Bs[2][BN * LDT];
  const int tid  = threadIdx.x;
  const int lane = tid & 31, wv = tid >> 5;
  const int half = lane >> 4, lr = lane & 15;
  const int wr = wv >> 1, wc = wv & 1;
  const size_t m0 = (size_t)blockIdx.y * BM;
  const size_t n0 = (size_t)blockIdx.x * BN;
  const int ar = tid >> 2, ac = (tid & 3) * 8;   // A staging coords
  const int br_ = tid >> 3, bc = (tid & 7) * 4;  // B staging coords

  f32x8 acc[2][4];
#pragma unroll
  for (int i = 0; i < 2; ++i)
#pragma unroll
    for (int j = 0; j < 4; ++j)
#pragma unroll
      for (int r = 0; r < 8; ++r) acc[i][j][r] = 0.f;

  f32x4 breg[4];
  // ---- prologue: stage K-tile 0 into buffer 0 ----
#pragma unroll
  for (int rr = 0; rr < BM; rr += 64)
    async_ld_b128(&As[0][(ar + rr) * LDT + ac], A + (m0 + ar + rr) * K + ac);
#pragma unroll
  for (int i = 0; i < 4; ++i)
    breg[i] = *(const f32x4*)(Bw + (n0 + br_ + 32 * i) * K + bc);
#pragma unroll
  for (int i = 0; i < 4; ++i) {
    u32x2 wpk;
    wpk.x = pack2bf(breg[i].x, breg[i].y);
    wpk.y = pack2bf(breg[i].z, breg[i].w);
    *(u32x2*)&Bs[0][(br_ + 32 * i) * LDT + bc] = wpk;
  }
  wait_async0();
  __syncthreads();

  const int KT = K / BK;
  for (int kt = 0; kt < KT; ++kt) {
    const int cur = kt & 1, nxt = cur ^ 1;
    const bool more = (kt + 1 < KT);
    if (more) {
      const int k1 = (kt + 1) * BK;
      // async A for next tile straight into the other LDS buffer
#pragma unroll
      for (int rr = 0; rr < BM; rr += 64)
        async_ld_b128(&As[nxt][(ar + rr) * LDT + ac],
                      A + (m0 + ar + rr) * K + k1 + ac);
      // B for next tile into registers (store to LDS after barrier #1)
#pragma unroll
      for (int i = 0; i < 4; ++i)
        breg[i] = *(const f32x4*)(Bw + (n0 + br_ + 32 * i) * K + k1 + bc);
      if (kt + 2 < KT)
        __builtin_prefetch((const void*)(Bw + (n0 + br_) * K + (kt + 2) * BK + bc), 0, 1);
    }

    // ---- compute on current buffers ----
    ABFrag af[2], bfr[4];
#pragma unroll
    for (int mf = 0; mf < 2; ++mf) {       // A frag: interleaved K layout
      const u16* p = &As[cur][(wr * 32 + mf * 16 + lr) * LDT + half * 8];
      af[mf].u[0] = *(const u32x4*)p;
      af[mf].u[1] = *(const u32x4*)(p + 16);
    }
#pragma unroll
    for (int nf = 0; nf < 4; ++nf) {       // B frag: interleaved K layout
      const u16* p = &Bs[cur][(wc * 64 + nf * 16 + lr) * LDT + half * 8];
      bfr[nf].u[0] = *(const u32x4*)p;
      bfr[nf].u[1] = *(const u32x4*)(p + 16);
    }
#pragma unroll
    for (int mf = 0; mf < 2; ++mf)
#pragma unroll
      for (int nf = 0; nf < 4; ++nf)
        acc[mf][nf] = wmma_bf16(af[mf], bfr[nf], acc[mf][nf]);

    __syncthreads();               // all waves done reading 'cur' (and old 'nxt')
    if (more) {
#pragma unroll
      for (int i = 0; i < 4; ++i) {
        u32x2 wpk;
        wpk.x = pack2bf(breg[i].x, breg[i].y);
        wpk.y = pack2bf(breg[i].z, breg[i].w);
        *(u32x2*)&Bs[nxt][(br_ + 32 * i) * LDT + bc] = wpk;
      }
    }
    wait_async0();                 // next A tile resident in LDS
    __syncthreads();               // publish next tile
  }

  // epilogue (C layout: VGPR r -> row r + half*8, col = lr)
#pragma unroll
  for (int mf = 0; mf < 2; ++mf) {
#pragma unroll
    for (int nf = 0; nf < 4; ++nf) {
      const size_t colg = n0 + wc * 64 + nf * 16 + lr;
      const float bv = bias ? bias[colg] : 0.f;
#pragma unroll
      for (int r = 0; r < 8; ++r) {
        const size_t rowg = m0 + wr * 32 + mf * 16 + half * 8 + r;
        float v = acc[mf][nf][r] + bv;
        if (mode == 1) {
          float* o = (float*)out + rowg * (size_t)N + colg;
          *o += v;
        } else if (mode == 2) {
          ((float*)out)[rowg * (size_t)N + colg] = v;
        } else {
          if (gelu) v = 0.5f * v * (1.0f + erff(v * 0.70710678118654752f));
          ((u16*)out)[rowg * (size_t)N + colg] = f2bf(v);
        }
      }
    }
  }
}

// ---------------------------------------------------------------------------
// Transpose V slice of qkv into vt[b,h,d,t] so attention B-operand loads are
// contiguous along kv.
// ---------------------------------------------------------------------------
__global__ __launch_bounds__(256) void vtrans_kernel(
    const u16* __restrict__ qkv, u16* __restrict__ vt)
{
  const size_t i = (size_t)blockIdx.x * 256 + threadIdx.x;  // < 4*NH*HDIM*LSEQ
  const int t = (int)(i & (LSEQ - 1));
  const int d = (int)((i >> 10) & (HDIM - 1));
  const int h = (int)((i >> 16) % NH);
  const int b = (int)((i >> 16) / NH);
  vt[i] = qkv[(size_t)(b * LSEQ + t) * (3 * CDIM) + 2 * CDIM + h * HDIM + d];
}

// ---------------------------------------------------------------------------
// Flash attention (causal), WMMA: each wave owns 16 q rows of one head.
// grid = (L/128, H, B); block = 256 (8 waves).
// ---------------------------------------------------------------------------
__global__ __launch_bounds__(256) void attn_kernel(
    const u16* __restrict__ qkv, const u16* __restrict__ vt,
    u16* __restrict__ y)
{
  __shared__ u16 plds[8 * 16 * LDT];
  const int qt = blockIdx.x, hh = blockIdx.y, bb = blockIdx.z;
  const int tid = threadIdx.x, lane = tid & 31, wv = tid >> 5;
  const int half = lane >> 4, lr = lane & 15;
  const int q0 = qt * 128 + wv * 16;
  const size_t rowg0 = (size_t)bb * LSEQ;

  // Q fragments (A layout), 16 rows x 64 dims = 2 frags of K=32
  ABFrag qa[2];
#pragma unroll
  for (int kc = 0; kc < 2; ++kc) {
    const u16* p = qkv + (rowg0 + q0 + lr) * (3 * CDIM)
                   + hh * HDIM + kc * 32 + half * 8;
    qa[kc].u[0] = *(const u32x4*)p;
    qa[kc].u[1] = *(const u32x4*)(p + 16);
  }

  f32x8 o[4];
  float m[8], l[8];
#pragma unroll
  for (int nf = 0; nf < 4; ++nf)
#pragma unroll
    for (int r = 0; r < 8; ++r) o[nf][r] = 0.f;
#pragma unroll
  for (int r = 0; r < 8; ++r) { m[r] = -1e30f; l[r] = 0.f; }

  u16* pl = &plds[wv * 16 * LDT];
  const int ntile = (q0 + 16 + 31) >> 5;   // 32-kv tiles covering causal span
  for (int j = 0; j < ntile; ++j) {
    const int kv0 = j * 32;
    // S = Q @ K^T for 32 kv columns (2 frags of 16)
    f32x8 s[2];
#pragma unroll
    for (int nt = 0; nt < 2; ++nt) {
#pragma unroll
      for (int r = 0; r < 8; ++r) s[nt][r] = 0.f;
#pragma unroll
      for (int kc = 0; kc < 2; ++kc) {
        ABFrag kf;  // B operand (interleaved layout) straight from K rows
        const u16* p = qkv + (rowg0 + kv0 + nt * 16 + lr) * (3 * CDIM)
                       + CDIM + hh * HDIM + kc * 32 + half * 8;
        kf.u[0] = *(const u32x4*)p;
        kf.u[1] = *(const u32x4*)(p + 16);
        s[nt] = wmma_bf16(qa[kc], kf, s[nt]);
      }
    }
    // scale + causal mask
#pragma unroll
    for (int nt = 0; nt < 2; ++nt)
#pragma unroll
      for (int r = 0; r < 8; ++r) {
        const int col = kv0 + nt * 16 + lr;
        const int row = q0 + half * 8 + r;
        const float v = s[nt][r] * 0.125f;   // 1/sqrt(64)
        s[nt][r] = (col > row) ? -1e30f : v;
      }
    // online softmax: row stats reduced across the 16-lane half
#pragma unroll
    for (int r = 0; r < 8; ++r) {
      float v = fmaxf(s[0][r], s[1][r]);
#pragma unroll
      for (int off = 1; off < 16; off <<= 1) v = fmaxf(v, __shfl_xor(v, off, 16));
      const float mn = fmaxf(m[r], v);
      const float alpha = __expf(m[r] - mn);
      m[r] = mn;
      const float p0 = __expf(s[0][r] - mn);
      const float p1 = __expf(s[1][r] - mn);
      s[0][r] = p0; s[1][r] = p1;
      float rs = p0 + p1;
#pragma unroll
      for (int off = 1; off < 16; off <<= 1) rs += __shfl_xor(rs, off, 16);
      l[r] = l[r] * alpha + rs;
#pragma unroll
      for (int nf = 0; nf < 4; ++nf) o[nf][r] *= alpha;
    }
    // P: C layout -> LDS -> A layout (per-wave region, DS ops are in-order)
#pragma unroll
    for (int r = 0; r < 8; ++r) {
      u32* prow = (u32*)&pl[(half * 8 + r) * LDT];   // lr pairs are not adjacent;
      (void)prow;                                    // keep scalar stores per col
      pl[(half * 8 + r) * LDT + 0 * 16 + lr] = f2bf(s[0][r]);
      pl[(half * 8 + r) * LDT + 1 * 16 + lr] = f2bf(s[1][r]);
    }
    ABFrag pa;
    {
      const u16* pp = &pl[lr * LDT + half * 8];
      pa.u[0] = *(const u32x4*)pp;
      pa.u[1] = *(const u32x4*)(pp + 16);
    }
    // O += P @ V  (V pre-transposed: contiguous along kv)
#pragma unroll
    for (int nf = 0; nf < 4; ++nf) {
      const u16* vp = vt + ((size_t)((bb * NH + hh) * HDIM) + nf * 16 + lr) * LSEQ
                      + kv0 + half * 8;
      ABFrag vb;
      vb.u[0] = *(const u32x4*)vp;
      vb.u[1] = *(const u32x4*)(vp + 16);
      o[nf] = wmma_bf16(pa, vb, o[nf]);
    }
  }
  // normalize and write y (bf16)
#pragma unroll
  for (int nf = 0; nf < 4; ++nf)
#pragma unroll
    for (int r = 0; r < 8; ++r) {
      const size_t rowg = rowg0 + q0 + half * 8 + r;
      const size_t colg = (size_t)hh * HDIM + nf * 16 + lr;
      y[rowg * CDIM + colg] = f2bf(o[nf][r] / l[r]);
    }
}

// ---------------------------------------------------------------------------
// Per-row log-softmax NLL over the 50304-wide logits
// ---------------------------------------------------------------------------
__global__ __launch_bounds__(256) void lossrow_kernel(
    const float* __restrict__ logits, const int* __restrict__ targets,
    float* __restrict__ nll, float* __restrict__ vcnt)
{
  __shared__ float red[256];
  const int row = blockIdx.x, tid = threadIdx.x;
  const float* lrow = logits + (size_t)row * VOC;
  float mx = -1e30f;
  for (int c = tid; c < VOC; c += 256) mx = fmaxf(mx, lrow[c]);
  red[tid] = mx; __syncthreads();
  for (int s2 = 128; s2 > 0; s2 >>= 1) {
    if (tid < s2) red[tid] = fmaxf(red[tid], red[tid + s2]);
    __syncthreads();
  }
  mx = red[0]; __syncthreads();
  float se = 0.f;
  for (int c = tid; c < VOC; c += 256) se += __expf(lrow[c] - mx);
  red[tid] = se; __syncthreads();
  for (int s2 = 128; s2 > 0; s2 >>= 1) {
    if (tid < s2) red[tid] += red[tid + s2];
    __syncthreads();
  }
  if (tid == 0) {
    const float lse = mx + __logf(red[0]);
    const int tg = targets[row];
    const bool valid = (tg != -1);
    const float lp = lrow[valid ? tg : 0] - lse;
    nll[row]  = valid ? -lp : 0.f;
    vcnt[row] = valid ? 1.f : 0.f;
  }
}

__global__ __launch_bounds__(256) void lossreduce_kernel(
    const float* __restrict__ nll, const float* __restrict__ vcnt,
    float* __restrict__ loss_out)
{
  __shared__ float sn[256], sv[256];
  const int tid = threadIdx.x;
  float a = 0.f, b = 0.f;
  for (int i = tid; i < TOK; i += 256) { a += nll[i]; b += vcnt[i]; }
  sn[tid] = a; sv[tid] = b; __syncthreads();
  for (int s2 = 128; s2 > 0; s2 >>= 1) {
    if (tid < s2) { sn[tid] += sn[tid + s2]; sv[tid] += sv[tid + s2]; }
    __syncthreads();
  }
  if (tid == 0) loss_out[0] = sn[0] / sv[0];
}

// ---------------------------------------------------------------------------
extern "C" void kernel_launch(void* const* d_in, const int* in_sizes, int n_in,
                              void* d_out, int out_size, void* d_ws, size_t ws_size,
                              hipStream_t stream)
{
  (void)in_sizes; (void)n_in; (void)out_size; (void)ws_size;
  const int*   idx     = (const int*)d_in[0];
  const int*   targets = (const int*)d_in[1];
  const float* wte     = (const float*)d_in[2];
  const float* wpe     = (const float*)d_in[3];
  const float* ln1_w   = (const float*)d_in[4];
  const float* ln1_b   = (const float*)d_in[5];
  const float* w_qkv   = (const float*)d_in[6];
  const float* b_qkv   = (const float*)d_in[7];
  const float* w_proj  = (const float*)d_in[8];
  const float* b_proj  = (const float*)d_in[9];
  const float* ln2_w   = (const float*)d_in[10];
  const float* ln2_b   = (const float*)d_in[11];
  const float* w_fc    = (const float*)d_in[12];
  const float* b_fc    = (const float*)d_in[13];
  const float* w_fc2   = (const float*)d_in[14];
  const float* b_fc2   = (const float*)d_in[15];
  const float* lnf_w   = (const float*)d_in[16];
  const float* lnf_b   = (const float*)d_in[17];

  char* ws = (char*)d_ws;
  float* x   = (float*)ws;  ws += (size_t)TOK * CDIM * 4;          // fp32 residual
  u16*   h   = (u16*)ws;    ws += (size_t)TOK * CDIM * 2;          // LN output bf16
  u16*   qkv = (u16*)ws;    ws += (size_t)TOK * 3 * CDIM * 2;      // QKV bf16
  u16*   vt  = (u16*)ws;    ws += (size_t)4 * NH * HDIM * LSEQ * 2;// V transposed
  u16*   y   = (u16*)ws;    ws += (size_t)TOK * CDIM * 2;          // attn out bf16
  u16*   fcb = (u16*)ws;    ws += (size_t)TOK * 4 * CDIM * 2;      // FC out bf16
  float* nll = (float*)ws;  ws += (size_t)TOK * 4;
  float* vcf = (float*)ws;  ws += (size_t)TOK * 4;

  float* logits = (float*)d_out;
  float* loss   = logits + (size_t)TOK * VOC;

  const dim3 blk(256);
  embed_kernel<<<TOK * CDIM / 256, blk, 0, stream>>>(idx, wte, wpe, x);

  for (int lyr = 0; lyr < NLAY; ++lyr) {
    ln_kernel<<<TOK / 8, blk, 0, stream>>>(x, ln1_w + lyr * CDIM, ln1_b + lyr * CDIM, h);
    gemm_kernel<<<dim3(3 * CDIM / BN, TOK / BM), blk, 0, stream>>>(
        h, w_qkv + (size_t)lyr * 3 * CDIM * CDIM, b_qkv + (size_t)lyr * 3 * CDIM,
        qkv, TOK, 3 * CDIM, CDIM, /*mode=*/0, /*gelu=*/0);
    vtrans_kernel<<<4 * NH * HDIM * LSEQ / 256, blk, 0, stream>>>(qkv, vt);
    attn_kernel<<<dim3(LSEQ / 128, NH, 4), blk, 0, stream>>>(qkv, vt, y);
    gemm_kernel<<<dim3(CDIM / BN, TOK / BM), blk, 0, stream>>>(
        y, w_proj + (size_t)lyr * CDIM * CDIM, b_proj + (size_t)lyr * CDIM,
        x, TOK, CDIM, CDIM, /*mode=*/1, /*gelu=*/0);
    ln_kernel<<<TOK / 8, blk, 0, stream>>>(x, ln2_w + lyr * CDIM, ln2_b + lyr * CDIM, h);
    gemm_kernel<<<dim3(4 * CDIM / BN, TOK / BM), blk, 0, stream>>>(
        h, w_fc + (size_t)lyr * 4 * CDIM * CDIM, b_fc + (size_t)lyr * 4 * CDIM,
        fcb, TOK, 4 * CDIM, CDIM, /*mode=*/0, /*gelu=*/1);
    gemm_kernel<<<dim3(CDIM / BN, TOK / BM), blk, 0, stream>>>(
        fcb, w_fc2 + (size_t)lyr * CDIM * 4 * CDIM, b_fc2 + (size_t)lyr * CDIM,
        x, TOK, CDIM, 4 * CDIM, /*mode=*/1, /*gelu=*/0);
  }

  ln_kernel<<<TOK / 8, blk, 0, stream>>>(x, lnf_w, lnf_b, h);
  gemm_kernel<<<dim3(VOC / BN, TOK / BM), blk, 0, stream>>>(
      h, wte, nullptr, logits, TOK, VOC, CDIM, /*mode=*/2, /*gelu=*/0);

  lossrow_kernel<<<TOK, blk, 0, stream>>>(logits, targets, nll, vcf);
  lossreduce_kernel<<<1, blk, 0, stream>>>(nll, vcf, loss);
}
